// GraphSAGELayer_55018531061841
// MI455X (gfx1250) — matrix operand order repaired and checked by reading, not compiled, not run on hardware
//
#include <hip/hip_runtime.h>
#include <hip/hip_bf16.h>

// ---------------------------------------------------------------------------
// GraphSAGE layer, MI455X (gfx1250, wave32, WMMA + TDM).
//   agg = (adj @ h) / N                 (big GEMM, K=2048, adj binary -> bf16 exact)
//   out = relu(h@Ws + agg@Wn + bs + bn) then row-wise L2 normalize
// B=8, N=2048, F_in=F_out=128.
// ---------------------------------------------------------------------------

typedef __attribute__((ext_vector_type(16))) __bf16       v16bf;
typedef __attribute__((ext_vector_type(8)))  float        v8f;
typedef __attribute__((ext_vector_type(4)))  unsigned int u32x4;
typedef __attribute__((ext_vector_type(8)))  int          i32x8;
typedef __attribute__((ext_vector_type(4)))  int          i32x4;
typedef __attribute__((ext_vector_type(4)))  float        f32x4;

union Frag16 { v16bf v; u32x4 u[2]; };

#define NB 8
#define NN 2048
#define NF 128
#define KC 64                    // K staged per barrier round (2 WMMA sub-steps)
#define ROWB 144                 // padded LDS row stride (128 B data + 16 B pad)
#define STAGEB (NF * ROWB)       // bytes per stage buffer = 18432

// ---------------------------------------------------------------------------
// Prep: h (f32, B x N x F) -> h_bf (row major bf16) and hT_bf (B x F x N bf16)
// ---------------------------------------------------------------------------
__global__ __launch_bounds__(256) void pack_h_kernel(const float* __restrict__ h,
                                                     __bf16* __restrict__ hbf,
                                                     __bf16* __restrict__ hT) {
  __shared__ float tile[32][33];
  const int b  = blockIdx.z;
  const int n0 = blockIdx.x * 32;
  const int f0 = blockIdx.y * 32;
#pragma unroll
  for (int j = 0; j < 4; ++j) {
    int n = n0 + threadIdx.y + j * 8;
    int f = f0 + threadIdx.x;
    size_t idx = ((size_t)b * NN + n) * NF + f;
    float v = h[idx];
    hbf[idx] = (__bf16)v;
    tile[threadIdx.y + j * 8][threadIdx.x] = v;
  }
  __syncthreads();
#pragma unroll
  for (int j = 0; j < 4; ++j) {
    int f = f0 + threadIdx.y + j * 8;
    int n = n0 + threadIdx.x;
    hT[((size_t)b * NF + f) * NN + n] = (__bf16)tile[threadIdx.x][threadIdx.y + j * 8];
  }
}

// W (F_in x F_out) f32 -> WT (F_out x F_in) bf16, for both weight matrices.
__global__ __launch_bounds__(256) void pack_w_kernel(const float* __restrict__ Ws,
                                                     const float* __restrict__ Wn,
                                                     __bf16* __restrict__ WsT,
                                                     __bf16* __restrict__ WnT) {
  int idx = blockIdx.x * 256 + threadIdx.x;   // 0 .. 16383
  int o = idx & (NF - 1);
  int i = idx >> 7;
  WsT[o * NF + i] = (__bf16)Ws[i * NF + o];
  WnT[o * NF + i] = (__bf16)Wn[i * NF + o];
}

// ---------------------------------------------------------------------------
// Main fused kernel. Block = 128 threads = 4 waves, each wave owns 16 rows.
// Grid = (N/64, B).
// LDS stage buffers hold hT tiles [128 rows f][64 K bf16] at a 144 B row
// stride (TDM pad: +4 dwords every 32 dwords) -> conflict-free half-wave
// ds_load_b128, 16 B aligned. agg rows padded to 136 bf16 (272 B).
// ---------------------------------------------------------------------------
struct LdsU {
  union {
    __bf16 stageT[2][STAGEB / 2];   // 2 x 18432 B = 36 KB
    __bf16 agg[4][16][136];         // per-wave agg tile, padded rows = 17 KB
  };
};

__global__ __launch_bounds__(128) void sage_main_kernel(
    const float*  __restrict__ adj,
    const __bf16* __restrict__ hbf,
    const __bf16* __restrict__ hT,
    const __bf16* __restrict__ WsT,
    const __bf16* __restrict__ WnT,
    const float*  __restrict__ bs,
    const float*  __restrict__ bn,
    float*        __restrict__ out) {
  __shared__ __align__(16) LdsU lds;

  const int b    = blockIdx.y;
  const int tid  = threadIdx.x;
  const int wave = tid >> 5;
  const int lane = tid & 31;
  const int hw   = lane >> 4;      // half-wave id
  const int l16  = lane & 15;
  const int m0   = blockIdx.x * 64 + wave * 16;   // first row of this wave's tile

  const float*  adjRow = adj + ((size_t)b * NN + (m0 + l16)) * NN;
  const __bf16* hTb    = hT + (size_t)b * NF * NN;

  v8f acc[8];
#pragma unroll
  for (int t = 0; t < 8; ++t) acc[t] = (v8f)0.0f;

#if __has_builtin(__builtin_amdgcn_tensor_load_to_lds)
  // --- Tensor Data Mover staging: one DMA per KC chunk, issued by wave 0 ---
  const unsigned stageBase = (unsigned)(uintptr_t)(void*)&lds.stageT[0][0];
  auto tdmStage = [&](int kblk, int buf) {
    unsigned long long ga =
        (unsigned long long)(uintptr_t)(const void*)(hTb + kblk);
    u32x4 g0;
    g0[0] = 1u;                                          // count=1, user mode
    g0[1] = stageBase + (unsigned)buf * STAGEB;          // lds_addr
    g0[2] = (unsigned)(ga & 0xFFFFFFFFu);                // global_addr[31:0]
    g0[3] = (unsigned)((ga >> 32) & 0x01FFFFFFu)         // global_addr[56:32]
            | (2u << 30);                                // type = 2 (image)
    i32x8 g1;
    g1[0] = (1 << 16)      // data_size = 2 bytes
          | (1 << 20)      // pad_enable
          | (4 << 22)      // pad_interval: every 32 dwords (= one 128 B row)
          | (3 << 25);     // pad_amount: 4 dwords (16 B) -> 144 B row stride
    g1[1] = (int)(2048u << 16);   // tensor_dim0 = 2048 (bits 79:48)
    g1[2] = (int)(128u << 16);    // tensor_dim1 = 128  (bits 111:80)
    g1[3] = (int)(64u << 16);     // tile_dim0   = 64   (bits 127:112)
    g1[4] = 128;                  // tile_dim1 = 128, tile_dim2 = 0
    g1[5] = 2048;                 // tensor_dim0_stride = 2048 (low 32)
    g1[6] = 0;
    g1[7] = 0;
    i32x4 z4 = (i32x4)0;          // D# groups 2/3 unused for 2D tile
    i32x8 z8 = (i32x8)0;
    __builtin_amdgcn_tensor_load_to_lds(g0, g1, z4, z4, z8, 0);
  };
#else
  // --- Fallback: cooperative copy into the identical padded layout ---------
  auto manualStage = [&](int kblk, int buf) {
    int f = tid;                                   // one row per thread
    const u32x4* src = (const u32x4*)(hTb + (size_t)f * NN + kblk);
    char* dst = (char*)&lds.stageT[buf][0] + f * ROWB;
#pragma unroll
    for (int j = 0; j < 8; ++j) *(u32x4*)(dst + j * 16) = src[j];
  };
#endif

  // Raw adj loads for one 32-wide K step (A fragment source, f32).
  auto loadA = [&](int kk, f32x4* a) {
    const int g0 = kk + hw * 16;
    const f32x4* p0 = (const f32x4*)(adjRow + g0);
    const f32x4* p1 = (const f32x4*)(adjRow + g0 + 8);
    a[0] = p0[0]; a[1] = p0[1]; a[2] = p1[0]; a[3] = p1[1];
  };

  f32x4 cur[4];
  loadA(0, cur);                    // software-pipelined adj stream

#if __has_builtin(__builtin_amdgcn_tensor_load_to_lds)
  if (wave == 0) {
    tdmStage(0, 0);
    __builtin_amdgcn_s_wait_tensorcnt(0);
  }
#else
  manualStage(0, 0);
#endif
  __syncthreads();

  // ---------------- big GEMM: acc = adj_tile @ h  (K = 2048) ----------------
  for (int k = 0; k < NN; k += KC) {
    const int buf = (k >> 6) & 1;
#if __has_builtin(__builtin_amdgcn_tensor_load_to_lds)
    if (wave == 0 && k + KC < NN) tdmStage(k + KC, buf ^ 1);
#else
    if (k + KC < NN) manualStage(k + KC, buf ^ 1);
#endif
#pragma unroll
    for (int ks = 0; ks < 2; ++ks) {
      const int kk = k + ks * 32;
      const int knext = (kk + 32 < NN) ? (kk + 32) : 0;   // branch-free clamp
      f32x4 nxt[4];
      loadA(knext, nxt);            // issue next step's adj loads early
      __builtin_prefetch((const void*)(adjRow + kk + 256), 0, 3);

      // A fragment: adj row (f32 0/1) -> bf16, exact.
      Frag16 fa;
#pragma unroll
      for (int e = 0; e < 4; ++e) {
        fa.v[e]      = (__bf16)cur[0][e];
        fa.v[4 + e]  = (__bf16)cur[1][e];
        fa.v[8 + e]  = (__bf16)cur[2][e];
        fa.v[12 + e] = (__bf16)cur[3][e];
      }

      // Load all 8 B fragments first so ds waits can stagger across WMMAs.
      const char* sb = (const char*)&lds.stageT[buf][0]
                     + (size_t)(ks * 32 + hw * 16) * 2 + (size_t)l16 * ROWB;
      Frag16 fb[8];
#pragma unroll
      for (int t = 0; t < 8; ++t) {
        const u32x4* p = (const u32x4*)(sb + (size_t)t * 16 * ROWB);
        fb[t].u[0] = p[0];
        fb[t].u[1] = p[1];
      }
#pragma unroll
      for (int t = 0; t < 8; ++t) {
        acc[t] = __builtin_amdgcn_wmma_f32_16x16x32_bf16(
            false, fa.v, false, fb[t].v, (short)0, acc[t], false, false);
      }
#pragma unroll
      for (int e = 0; e < 4; ++e) cur[e] = nxt[e];
    }
#if __has_builtin(__builtin_amdgcn_tensor_load_to_lds)
    if (wave == 0 && k + KC < NN) __builtin_amdgcn_s_wait_tensorcnt(0);
#endif
    __syncthreads();
  }

  // ------------- spill agg (= acc/N) to LDS as bf16, A-loadable -------------
  const float invN = 1.0f / (float)NN;
#pragma unroll
  for (int t = 0; t < 8; ++t) {
#pragma unroll
    for (int r = 0; r < 8; ++r) {
      int row = r + hw * 8;            // C layout: VGPR r holds rows r / r+8
      int col = t * 16 + l16;
      lds.agg[wave][row][col] = (__bf16)(acc[t][r] * invN);
    }
  }
  __syncthreads();

  // -------- small GEMMs: o = h@Ws + agg@Wn  (K = 128, 4 WMMA steps) --------
  const __bf16* hrow = hbf + ((size_t)b * NN + m0 + l16) * NF;
  v8f o2[8];
#pragma unroll
  for (int t = 0; t < 8; ++t) o2[t] = (v8f)0.0f;

#pragma unroll
  for (int kk = 0; kk < NF; kk += 32) {
    const int g0 = kk + hw * 16;
    Frag16 fh, fg;
    fh.u[0] = *(const u32x4*)(hrow + g0);
    fh.u[1] = *(const u32x4*)(hrow + g0 + 8);
    fg.u[0] = *(const u32x4*)&lds.agg[wave][l16][g0];
    fg.u[1] = *(const u32x4*)&lds.agg[wave][l16][g0 + 8];
#pragma unroll
    for (int t = 0; t < 8; ++t) {
      int o = t * 16 + l16;            // B layout: out-feature per lane
      Frag16 fw, fn;
      const u32x4* pw = (const u32x4*)(WsT + o * NF + kk + hw * 16);
      fw.u[0] = pw[0]; fw.u[1] = pw[1];
      o2[t] = __builtin_amdgcn_wmma_f32_16x16x32_bf16(
          false, fh.v, false, fw.v, (short)0, o2[t], false, false);
      const u32x4* pn = (const u32x4*)(WnT + o * NF + kk + hw * 16);
      fn.u[0] = pn[0]; fn.u[1] = pn[1];
      o2[t] = __builtin_amdgcn_wmma_f32_16x16x32_bf16(
          false, fg.v, false, fn.v, (short)0, o2[t], false, false);
    }
  }

  // ---------------------- bias + ReLU in C layout ---------------------------
#pragma unroll
  for (int t = 0; t < 8; ++t) {
    int col = t * 16 + l16;
    float bias = bs[col] + bn[col];
#pragma unroll
    for (int r = 0; r < 8; ++r) {
      float v = o2[t][r] + bias;
      o2[t][r] = v > 0.0f ? v : 0.0f;
    }
  }

  // ------------- row L2 norm: half-wave shuffle reduction -------------------
  float rinv[8];
#pragma unroll
  for (int r = 0; r < 8; ++r) {
    float ss = 0.0f;
#pragma unroll
    for (int t = 0; t < 8; ++t) ss += o2[t][r] * o2[t][r];
    ss += __shfl_xor(ss, 1);
    ss += __shfl_xor(ss, 2);
    ss += __shfl_xor(ss, 4);
    ss += __shfl_xor(ss, 8);           // masks <=8 stay within each 16-lane half
    float nrm = sqrtf(ss);
    rinv[r] = 1.0f / fmaxf(nrm, 1e-12f);
  }

  // ----------------------------- store --------------------------------------
  float* outp = out + ((size_t)b * NN + m0) * NF;
#pragma unroll
  for (int t = 0; t < 8; ++t) {
#pragma unroll
    for (int r = 0; r < 8; ++r) {
      int row = r + hw * 8;
      outp[(size_t)row * NF + t * 16 + l16] = o2[t][r] * rinv[r];
    }
  }
}

// ---------------------------------------------------------------------------
extern "C" void kernel_launch(void* const* d_in, const int* in_sizes, int n_in,
                              void* d_out, int out_size, void* d_ws, size_t ws_size,
                              hipStream_t stream) {
  const float* h   = (const float*)d_in[0];   // (8,2048,128)
  const float* adj = (const float*)d_in[1];   // (8,2048,2048)
  const float* Ws  = (const float*)d_in[2];   // (128,128)
  const float* bsv = (const float*)d_in[3];   // (128,)
  const float* Wn  = (const float*)d_in[4];   // (128,128)
  const float* bnv = (const float*)d_in[5];   // (128,)
  float* out = (float*)d_out;

  char* ws = (char*)d_ws;
  const size_t hbytes = (size_t)NB * NN * NF * sizeof(__bf16);   // 4 MiB
  __bf16* hbf  = (__bf16*)(ws);
  __bf16* hT   = (__bf16*)(ws + hbytes);
  __bf16* WsT  = (__bf16*)(ws + 2 * hbytes);
  __bf16* WnT  = (__bf16*)(ws + 2 * hbytes + (size_t)NF * NF * sizeof(__bf16));

  // prep: convert/transpose h and weights to bf16
  pack_h_kernel<<<dim3(NN / 32, NF / 32, NB), dim3(32, 8), 0, stream>>>(h, hbf, hT);
  pack_w_kernel<<<dim3((NF * NF) / 256), dim3(256), 0, stream>>>(Ws, Wn, WsT, WnT);

  // main fused kernel: 32 row-blocks x 8 batches = 256 workgroups
  sage_main_kernel<<<dim3(NN / 64, NB), dim3(128), 0, stream>>>(
      adj, hbf, hT, WsT, WnT, bsv, bnv, out);
}
